// UpDownCaptionerAttention_13615046328944
// MI455X (gfx1250) — compile-verified, matrix operand order checked by reading, not run on hardware
//
#include <hip/hip_runtime.h>
#include <hip/hip_bf16.h>
#include <math.h>

typedef __attribute__((ext_vector_type(16))) __bf16 v16bf;
typedef __attribute__((ext_vector_type(8)))  __bf16 v8bf;
typedef __attribute__((ext_vector_type(4)))  __bf16 v4bf;
typedef __attribute__((ext_vector_type(8)))  float  v8f;

#define FEAT 2048
#define HID  1024
#define ATT  512
#define BB   256
#define RR   196
#define RT   13              // ceil(196/16) row tiles
#define KH   1024            // K half staged in LDS
#define LDA  (KH + 16)       // padded LDS row stride (bf16 elems): +32B -> 8-bank row skew
#define NEGV (-1000000000.0f)

#if __has_builtin(__builtin_amdgcn_tanhf)
__device__ __forceinline__ float fast_tanh(float x) { return __builtin_amdgcn_tanhf(x); }
#else
__device__ __forceinline__ float fast_tanh(float x) { return tanhf(x); }
#endif

// Load 16 contiguous f32 (64B) and convert to a bf16 WMMA fragment half-row.
__device__ __forceinline__ v16bf load16_cvt(const float* __restrict__ p) {
  const float4* q = (const float4*)p;
  float4 x0 = q[0], x1 = q[1], x2 = q[2], x3 = q[3];
  v16bf r;
  r[0]  = (__bf16)x0.x; r[1]  = (__bf16)x0.y; r[2]  = (__bf16)x0.z; r[3]  = (__bf16)x0.w;
  r[4]  = (__bf16)x1.x; r[5]  = (__bf16)x1.y; r[6]  = (__bf16)x1.z; r[7]  = (__bf16)x1.w;
  r[8]  = (__bf16)x2.x; r[9]  = (__bf16)x2.y; r[10] = (__bf16)x2.z; r[11] = (__bf16)x2.w;
  r[12] = (__bf16)x3.x; r[13] = (__bf16)x3.y; r[14] = (__bf16)x3.z; r[15] = (__bf16)x3.w;
  return r;
}

// Kernel 0: W_f f32 -> bf16 (one-shot, 2 MB)
__global__ __launch_bounds__(256) void cvt_wf_kernel(
    const float* __restrict__ W_f, __bf16* __restrict__ out) {
  const int i = blockIdx.x * 256 + threadIdx.x;   // grid covers ATT*FEAT/4
  float4 v = ((const float4*)W_f)[i];
  v4bf o;
  o[0] = (__bf16)v.x; o[1] = (__bf16)v.y; o[2] = (__bf16)v.z; o[3] = (__bf16)v.w;
  ((v4bf*)out)[i] = o;
}

// Kernel 1: c2[b,a] = dot(hidden[b,:], W_h[a,:]) + b_h[a] + b_f[a]
__global__ __launch_bounds__(256) void attn_att2_kernel(
    const float* __restrict__ hidden, const float* __restrict__ W_h,
    const float* __restrict__ b_h, const float* __restrict__ b_f,
    float* __restrict__ c2) {
  const int b = blockIdx.x;
  const int t = threadIdx.x;
  __shared__ float h_lds[HID];
  for (int i = t; i < HID; i += 256) h_lds[i] = hidden[b * HID + i];
  __syncthreads();
  for (int a = t; a < ATT; a += 256) {
    const float4* wr = (const float4*)(W_h + (size_t)a * HID);
    float acc = b_h[a] + b_f[a];
    #pragma unroll 4
    for (int h4 = 0; h4 < HID / 4; ++h4) {
      float4 w = wr[h4];
      int h = h4 * 4;
      acc = fmaf(h_lds[h + 0], w.x, acc);
      acc = fmaf(h_lds[h + 1], w.y, acc);
      acc = fmaf(h_lds[h + 2], w.z, acc);
      acc = fmaf(h_lds[h + 3], w.w, acc);
    }
    c2[b * ATT + a] = acc;
  }
}

// Kernel 2: one block per batch. LDS-staged bf16 A tiles + WMMA att1,
// fused tanh/score, masked softmax, context GEMV.
template <bool WBF>
__global__ __launch_bounds__(256) void attn_main_kernel(
    const float* __restrict__ features, const unsigned char* __restrict__ mask,
    const float* __restrict__ W_f, const __bf16* __restrict__ Wf_bf,
    const float* __restrict__ W_a, const float* __restrict__ b_a,
    const float* __restrict__ c2,
    float* __restrict__ out_ctx, float* __restrict__ out_alpha) {
  const int b    = blockIdx.x;
  const int t    = threadIdx.x;
  const int wid  = t >> 5;
  const int lane = t & 31;

  __shared__ __bf16 lds_A[16 * LDA];      // 33.3 KB staged A half-tile (bf16)
  __shared__ float c2_lds[ATT];
  __shared__ float wa_lds[ATT];
  __shared__ float e_lds[RT * 16];
  __shared__ float alpha_lds[RT * 16];
  __shared__ float red[8];
  __shared__ float bcast[2];

  for (int i = t; i < ATT; i += 256) {
    c2_lds[i] = c2[b * ATT + i];
    wa_lds[i] = W_a[i];
  }
  for (int i = t; i < RT * 16; i += 256) e_lds[i] = 0.0f;
  const float ba0 = b_a[0];
  __syncthreads();

  const float* featb = features + (size_t)b * RR * FEAT;

  // ---- Pass 1: wave owns a fixed 64-wide a-slice (atg = wid); loop r-tiles -
  const int n16  = lane & 15;          // M (A) / N (B,D) index within tile
  const int koff = (lane >> 4) << 4;   // 0 or 16: per-lane K chunk
  const int a0g  = wid << 6;           // this wave's first 'a'

  // Hoist per-wave B pointers and score constants (fixed a-slice).
  const float*  wf32[4];
  const __bf16* wb16[4];
  float c2a[4], waa[4];
  #pragma unroll
  for (int j = 0; j < 4; ++j) {
    const int a = a0g + j * 16 + n16;
    const size_t rowoff = (size_t)a * FEAT + koff;
    wf32[j] = W_f + rowoff;
    wb16[j] = Wf_bf + rowoff;
    c2a[j]  = c2_lds[a];
    waa[j]  = wa_lds[a];
  }

  for (int rt = 0; rt < RT; ++rt) {
    const int r0 = rt << 4;

    v8f acc[4];
    #pragma unroll
    for (int j = 0; j < 4; ++j) acc[j] = (v8f){0.f,0.f,0.f,0.f,0.f,0.f,0.f,0.f};

    for (int kh = 0; kh < FEAT; kh += KH) {
      // -- cooperative stage: features[r0..r0+15, kh..kh+KH) -> bf16 in LDS --
      __syncthreads();
      for (int idx = t; idx < 16 * (KH / 8); idx += 256) {   // 8 f32 per chunk
        const int row  = idx >> 7;                           // KH/8 = 128 chunks/row
        const int col8 = (idx & 127) << 3;
        int gr = r0 + row;
        if (gr > RR - 1) gr = RR - 1;                        // clamp (pad rows)
        const float4* src = (const float4*)(featb + (size_t)gr * FEAT + kh + col8);
        float4 a0 = src[0], a1 = src[1];
        v8bf o;
        o[0] = (__bf16)a0.x; o[1] = (__bf16)a0.y; o[2] = (__bf16)a0.z; o[3] = (__bf16)a0.w;
        o[4] = (__bf16)a1.x; o[5] = (__bf16)a1.y; o[6] = (__bf16)a1.z; o[7] = (__bf16)a1.w;
        *(v8bf*)&lds_A[row * LDA + col8] = o;
      }
      __syncthreads();

      // -- WMMA over this K half: A from LDS, B streamed from global --------
      const __bf16* arow = &lds_A[n16 * LDA + koff];
      #pragma unroll 2
      for (int k0 = 0; k0 < KH; k0 += 32) {
        v16bf af = *(const v16bf*)(arow + k0);
        #pragma unroll
        for (int j = 0; j < 4; ++j) {
          v16bf bfj = WBF ? *(const v16bf*)(wb16[j] + kh + k0)
                          : load16_cvt(wf32[j] + kh + k0);
          acc[j] = __builtin_amdgcn_wmma_f32_16x16x32_bf16(
              false, af, false, bfj, (short)0, acc[j], false, false);
        }
      }
    }

    // -- fused score: e[r] += sum_a tanh(att1 + c2[a]) * W_a[a] -------------
    const int rbase = r0 + ((lane >> 4) << 3);
    #pragma unroll
    for (int j = 0; j < 4; ++j) {
      #pragma unroll
      for (int g = 0; g < 8; ++g) {
        float v = fast_tanh(acc[j][g] + c2a[j]) * waa[j];
        v += __shfl_xor(v, 1, 32);
        v += __shfl_xor(v, 2, 32);
        v += __shfl_xor(v, 4, 32);
        v += __shfl_xor(v, 8, 32);
        const int r = rbase + g;
        if ((lane & 15) == 0 && r < RR) atomicAdd(&e_lds[r], v);
      }
    }
  }
  __syncthreads();

  // ---------------- Softmax over R with mask ------------------------------
  float x = -3.0e38f;
  if (t < RR) {
    const bool mk = mask[b * RR + t] != 0;
    x = mk ? (e_lds[t] + ba0) : NEGV;
  }
  float m = x;
  m = fmaxf(m, __shfl_xor(m, 16, 32));
  m = fmaxf(m, __shfl_xor(m, 8, 32));
  m = fmaxf(m, __shfl_xor(m, 4, 32));
  m = fmaxf(m, __shfl_xor(m, 2, 32));
  m = fmaxf(m, __shfl_xor(m, 1, 32));
  if (lane == 0) red[wid] = m;
  __syncthreads();
  if (t == 0) {
    float mm = red[0];
    for (int i = 1; i < 8; ++i) mm = fmaxf(mm, red[i]);
    bcast[0] = mm;
  }
  __syncthreads();
  m = bcast[0];

  float ex = (t < RR) ? __expf(x - m) : 0.0f;
  float s = ex;
  s += __shfl_xor(s, 16, 32);
  s += __shfl_xor(s, 8, 32);
  s += __shfl_xor(s, 4, 32);
  s += __shfl_xor(s, 2, 32);
  s += __shfl_xor(s, 1, 32);
  if (lane == 0) red[wid] = s;
  __syncthreads();
  if (t == 0) {
    float ss = 0.f;
    for (int i = 0; i < 8; ++i) ss += red[i];
    bcast[1] = ss;
  }
  __syncthreads();
  const float inv = 1.0f / bcast[1];
  if (t < RR) {
    const float al = ex * inv;
    alpha_lds[t] = al;
    out_alpha[b * RR + t] = al;
  }
  __syncthreads();

  // ---------------- Pass 2: context[b,f] = sum_r alpha[r]*features[b,r,f] --
  #pragma unroll
  for (int j = 0; j < FEAT / 256; ++j) {
    const int f = t + j * 256;
    float acc = 0.0f;
    #pragma unroll 4
    for (int r = 0; r < RR; ++r)
      acc = fmaf(alpha_lds[r], featb[(size_t)r * FEAT + f], acc);
    out_ctx[(size_t)b * FEAT + f] = acc;
  }
}

extern "C" void kernel_launch(void* const* d_in, const int* in_sizes, int n_in,
                              void* d_out, int out_size, void* d_ws, size_t ws_size,
                              hipStream_t stream) {
  const float*         features = (const float*)d_in[0];
  const float*         hidden   = (const float*)d_in[1];
  const unsigned char* fmask    = (const unsigned char*)d_in[2];
  const float*         W_f      = (const float*)d_in[3];
  const float*         b_f      = (const float*)d_in[4];
  const float*         W_h      = (const float*)d_in[5];
  const float*         b_h      = (const float*)d_in[6];
  const float*         W_a      = (const float*)d_in[7];
  const float*         b_a      = (const float*)d_in[8];

  float* out_ctx   = (float*)d_out;                 // [B, FEAT]
  float* out_alpha = out_ctx + (size_t)BB * FEAT;   // [B, RR]

  const size_t c2_bytes = (size_t)BB * ATT * sizeof(float);       // 512 KB
  const size_t wf_bytes = (size_t)ATT * FEAT * sizeof(__bf16);    // 2 MB
  float*  c2    = (float*)d_ws;
  __bf16* Wf_bf = (__bf16*)((char*)d_ws + c2_bytes);
  const bool use_bf = (ws_size >= c2_bytes + wf_bytes);

  attn_att2_kernel<<<BB, 256, 0, stream>>>(hidden, W_h, b_h, b_f, c2);
  if (use_bf) {
    cvt_wf_kernel<<<(ATT * FEAT / 4) / 256, 256, 0, stream>>>(W_f, Wf_bf);
    attn_main_kernel<true><<<BB, 256, 0, stream>>>(
        features, fmask, W_f, Wf_bf, W_a, b_a, c2, out_ctx, out_alpha);
  } else {
    attn_main_kernel<false><<<BB, 256, 0, stream>>>(
        features, fmask, W_f, Wf_bf, W_a, b_a, c2, out_ctx, out_alpha);
  }
}
